// TinyFractal_1159641170579
// MI455X (gfx1250) — matrix operand order repaired and checked by reading, not compiled, not run on hardware
//
#include <hip/hip_runtime.h>

// ---------------------------------------------------------------------------
// Model dims (fixed by the reference)
// ---------------------------------------------------------------------------
#define B_ 4
#define T_ 1024
#define D_ 2048
#define H_ 16
#define HD_ 128
#define DFF_ 8192
#define DIN_ 4096
#define DCONV_ 4
#define VOCAB_ 32000
#define BT_ (B_ * T_)

// ---------------------------------------------------------------------------
// Types for CDNA5 WMMA (wave32)
// ---------------------------------------------------------------------------
typedef __attribute__((ext_vector_type(16))) __bf16 v16bf;
typedef __attribute__((ext_vector_type(8)))  float  v8f;

union FragBF {
    uint4 q[2];
    v16bf v;
};

__device__ __forceinline__ unsigned short f2bf(float f) {
    union { float f; unsigned int u; } cv;
    cv.f = f;
    unsigned int u = cv.u;
    u += 0x7FFFu + ((u >> 16) & 1u);   // round-to-nearest-even
    return (unsigned short)(u >> 16);
}

__device__ __forceinline__ float silu_f(float x) {
    return x / (1.0f + __expf(-x));
}

__device__ __forceinline__ void wait_asynccnt0() {
#if __has_builtin(__builtin_amdgcn_s_wait_asynccnt)
    __builtin_amdgcn_s_wait_asynccnt(0);
#else
    asm volatile("s_wait_asynccnt 0x0" ::: "memory");
#endif
}

// Issue GLOBAL_LOAD_ASYNC_TO_LDS_B128: 16B per lane, global -> LDS, ASYNCcnt.
__device__ __forceinline__ void async_copy16(unsigned lds_off, const void* gptr) {
    unsigned long long ga = (unsigned long long)(uintptr_t)gptr;
    asm volatile("global_load_async_to_lds_b128 %0, %1, off"
                 :: "v"(lds_off), "v"(ga) : "memory");
}

// ---------------------------------------------------------------------------
// fp32 -> bf16 bulk convert (vectorized by 4)
// ---------------------------------------------------------------------------
__global__ __launch_bounds__(256) void k_cvt_bf16(const float* __restrict__ s,
                                                  unsigned short* __restrict__ d,
                                                  long long n4) {
    long long i = (long long)blockIdx.x * 256 + threadIdx.x;
    if (i >= n4) return;
    float4 f = reinterpret_cast<const float4*>(s)[i];
    uint2 p;
    p.x = (unsigned)f2bf(f.x) | ((unsigned)f2bf(f.y) << 16);
    p.y = (unsigned)f2bf(f.z) | ((unsigned)f2bf(f.w) << 16);
    reinterpret_cast<uint2*>(d)[i] = p;
}

// ---------------------------------------------------------------------------
// Embedding gather: x[bt, :] = embed_w[idx[bt], :]
// ---------------------------------------------------------------------------
__global__ __launch_bounds__(256) void k_gather(const int* __restrict__ idx,
                                                const float* __restrict__ E,
                                                float* __restrict__ x) {
    int bt = blockIdx.x;
    long long row = idx[bt];
    const float4* src = reinterpret_cast<const float4*>(E + row * (long long)D_);
    float4* dst = reinterpret_cast<float4*>(x + (long long)bt * D_);
    for (int i = threadIdx.x; i < D_ / 4; i += 256) dst[i] = src[i];
}

// ---------------------------------------------------------------------------
// Tiled bf16 WMMA GEMM:  C[M,N] (+)= epi( alpha * A[M,K] * B[N,K]^T + bias[N] )
//   A: fp32 (loaded to regs, converted to bf16, stored to LDS)
//   B: bf16 (row-major [N,K]) moved global->LDS with GLOBAL_LOAD_ASYNC_TO_LDS_B128
// Block: 256 threads = 8 waves; tile 128x128, K-step 32.
// Wave (wm in 0..3, wn in 0..1) owns a 32x64 sub-tile = 2x4 WMMA accumulators.
// EPI: 0=store, 1=add-into-C (residual), 2=relu^2 store, 3=softplus store
// Batched via z: off = (z/batchH)*s?b + (z%batchH)*s?h
// ---------------------------------------------------------------------------
#define BK 32
#define LDS_K 40   // padded LDS row stride in halves (80B) to dodge bank conflicts

template <int EPI>
__global__ __launch_bounds__(256, 1)
void k_gemm(const float* __restrict__ A, const unsigned short* __restrict__ Bw,
            float* __restrict__ C, const float* __restrict__ bias,
            int M, int N, int K, int lda, int ldb, int ldc,
            long long sAb, long long sAh, long long sBb, long long sBh,
            long long sCb, long long sCh, int batchH, float alpha) {
    __shared__ alignas(16) unsigned short As[128 * LDS_K];
    __shared__ alignas(16) unsigned short Bs[128 * LDS_K];

    const int tid  = threadIdx.x;
    const int wave = tid >> 5, lane = tid & 31;
    const int g = lane >> 4, l15 = lane & 15;
    const int wm = wave & 3, wn = wave >> 2;

    const long long z  = blockIdx.z;
    const long long bb = z / batchH, hh = z % batchH;
    A  += bb * sAb + hh * sAh;
    Bw += bb * sBb + hh * sBh;
    C  += bb * sCb + hh * sCh;

    const int blockN = blockIdx.x * 128;
    const int blockM = blockIdx.y * 128;

    v8f acc[2][4] = {};

    const int ar = tid >> 3;          // 0..31
    const int ac = (tid & 7) * 4;     // 0,4,..,28
    const int br = tid >> 2;          // 0..63
    const int bc = (tid & 3) * 8;     // 0,8,16,24

    // LDS byte offsets for this thread's async B destinations
    const unsigned bsBase = (unsigned)(uintptr_t)(&Bs[0]);
    const unsigned bLds0 = bsBase + (unsigned)(((br      ) * LDS_K + bc) * 2);
    const unsigned bLds1 = bsBase + (unsigned)(((br + 64 ) * LDS_K + bc) * 2);

    for (int k0 = 0; k0 < K; k0 += BK) {
        // ---- async B tile: global -> LDS directly (ASYNCcnt) ----
        async_copy16(bLds0, &Bw[(long long)(blockN + br     ) * ldb + k0 + bc]);
        async_copy16(bLds1, &Bw[(long long)(blockN + br + 64) * ldb + k0 + bc]);

        // ---- A tile: issue all loads first (keep them in flight together) ----
        float4 aR[4];
#pragma unroll
        for (int p = 0; p < 4; ++p)
            aR[p] = *reinterpret_cast<const float4*>(
                &A[(long long)(blockM + ar + p * 32) * lda + k0 + ac]);

        if (k0 + BK < K)
            __builtin_prefetch(&A[(long long)(blockM + ar) * lda + k0 + BK + ac], 0, 3);

        // ---- convert fp32 -> bf16 and store A tile to LDS ----
#pragma unroll
        for (int p = 0; p < 4; ++p) {
            unsigned short* dst = &As[(ar + p * 32) * LDS_K + ac];
            dst[0] = f2bf(aR[p].x); dst[1] = f2bf(aR[p].y);
            dst[2] = f2bf(aR[p].z); dst[3] = f2bf(aR[p].w);
        }

        // Publish: my async B writes + my ds stores complete, then block barrier.
        wait_asynccnt0();
        __syncthreads();

        // ---- A fragments (16x32 bf16 layout: lane-half g holds K [8g,8g+8)+[16+8g,..)) ----
        FragBF fa[2];
#pragma unroll
        for (int mt = 0; mt < 2; ++mt) {
            int row = wm * 32 + mt * 16 + l15;
            fa[mt].q[0] = *reinterpret_cast<const uint4*>(&As[row * LDS_K + g * 8]);
            fa[mt].q[1] = *reinterpret_cast<const uint4*>(&As[row * LDS_K + 16 + g * 8]);
        }
        // ---- B fragments (32x16: lane-half g holds K [8g,8g+8)+[16+8g,..)) + 8 WMMAs ----
#pragma unroll
        for (int nt = 0; nt < 4; ++nt) {
            int col = wn * 64 + nt * 16 + l15;
            FragBF fb;
            fb.q[0] = *reinterpret_cast<const uint4*>(&Bs[col * LDS_K + g * 8]);
            fb.q[1] = *reinterpret_cast<const uint4*>(&Bs[col * LDS_K + 16 + g * 8]);
#pragma unroll
            for (int mt = 0; mt < 2; ++mt) {
                acc[mt][nt] = __builtin_amdgcn_wmma_f32_16x16x32_bf16(
                    false, fa[mt].v, false, fb.v, (short)0, acc[mt][nt], false, false);
            }
        }
        __syncthreads();
    }

    // ---- epilogue (C layout: VGPR r -> row 8g+r, col l15) ----
#pragma unroll
    for (int mt = 0; mt < 2; ++mt) {
#pragma unroll
        for (int nt = 0; nt < 4; ++nt) {
            int col = blockN + wn * 64 + nt * 16 + l15;
            float bv = bias ? bias[col] : 0.0f;
#pragma unroll
            for (int r = 0; r < 8; ++r) {
                int row = blockM + wm * 32 + mt * 16 + g * 8 + r;
                float v = acc[mt][nt][r] * alpha + bv;
                float* cp = &C[(long long)row * ldc + col];
                if (EPI == 0) {
                    *cp = v;
                } else if (EPI == 1) {
                    *cp += v;
                } else if (EPI == 2) {
                    float rl = fmaxf(v, 0.0f);
                    *cp = rl * rl;
                } else {
                    *cp = (v > 20.0f) ? v : __logf(1.0f + __expf(v));
                }
            }
        }
    }
}

// ---------------------------------------------------------------------------
// Causal softmax over scores[bh, q, :] (len = q+1); zero the masked tail so the
// following GEMM can consume the full K=T row.
// ---------------------------------------------------------------------------
__global__ __launch_bounds__(256) void k_softmax(float* __restrict__ S) {
    const int q = blockIdx.x, bh = blockIdx.y;
    float* row = S + ((long long)bh * T_ + q) * (long long)T_;
    const int len = q + 1;
    const int tid = threadIdx.x;
    __shared__ float red[8];

    float mx = -3.4e38f;
    for (int i = tid; i < len; i += 256) mx = fmaxf(mx, row[i]);
#pragma unroll
    for (int o = 16; o > 0; o >>= 1) mx = fmaxf(mx, __shfl_xor(mx, o, 32));
    if ((tid & 31) == 0) red[tid >> 5] = mx;
    __syncthreads();
    mx = red[0];
#pragma unroll
    for (int w = 1; w < 8; ++w) mx = fmaxf(mx, red[w]);
    __syncthreads();

    float sum = 0.0f;
    for (int i = tid; i < len; i += 256) {
        float e = __expf(row[i] - mx);
        row[i] = e;
        sum += e;
    }
#pragma unroll
    for (int o = 16; o > 0; o >>= 1) sum += __shfl_xor(sum, o, 32);
    if ((tid & 31) == 0) red[tid >> 5] = sum;
    __syncthreads();
    sum = red[0];
#pragma unroll
    for (int w = 1; w < 8; ++w) sum += red[w];
    float inv = 1.0f / sum;

    for (int i = tid; i < T_; i += 256) row[i] = (i < len) ? row[i] * inv : 0.0f;
}

// ---------------------------------------------------------------------------
// V transpose: v[b,t,h,hd] (fp32) -> vt[b,h,hd,t] (bf16) for the S*V GEMM
// ---------------------------------------------------------------------------
__global__ __launch_bounds__(256) void k_transpose_v(const float* __restrict__ v,
                                                     unsigned short* __restrict__ vt) {
    long long i = (long long)blockIdx.x * 256 + threadIdx.x;
    if (i >= (long long)B_ * H_ * HD_ * T_) return;
    int t = (int)(i % T_);
    long long r = i / T_;
    int hd = (int)(r % HD_);
    long long r2 = r / HD_;
    int h = (int)(r2 % H_);
    int b = (int)(r2 / H_);
    vt[i] = f2bf(v[((long long)(b * T_ + t)) * D_ + h * HD_ + hd]);
}

// ---------------------------------------------------------------------------
// Depthwise causal conv (left pad 3) + bias + SiLU over x_proj = xz[..., :DIN]
// ---------------------------------------------------------------------------
__global__ __launch_bounds__(256) void k_conv_silu(const float* __restrict__ xz,
                                                   const float* __restrict__ cw,
                                                   const float* __restrict__ cb,
                                                   float* __restrict__ xconv) {
    long long i = (long long)blockIdx.x * 256 + threadIdx.x;
    if (i >= (long long)BT_ * DIN_) return;
    int c = (int)(i % DIN_);
    long long bt = i / DIN_;
    int t = (int)(bt % T_);
    long long brow = (bt - t);  // b*T
    float s = cb[c];
#pragma unroll
    for (int j = 0; j < DCONV_; ++j) {
        int tt = t - (DCONV_ - 1) + j;
        if (tt >= 0)
            s += cw[c * DCONV_ + j] * xz[(brow + tt) * (2LL * DIN_) + c];
    }
    xconv[i] = silu_f(s);
}

// ---------------------------------------------------------------------------
// Mamba mix: y = xconv * (dt + D_param) * silu(z)
// ---------------------------------------------------------------------------
__global__ __launch_bounds__(256) void k_mamba_mix(const float* __restrict__ xconv,
                                                   const float* __restrict__ dt,
                                                   const float* __restrict__ xz,
                                                   const float* __restrict__ Dp,
                                                   float* __restrict__ y) {
    long long i = (long long)blockIdx.x * 256 + threadIdx.x;
    if (i >= (long long)BT_ * DIN_) return;
    int c = (int)(i % DIN_);
    long long bt = i / DIN_;
    float z = xz[bt * (2LL * DIN_) + DIN_ + c];
    float v = xconv[i] * (dt[i] + Dp[c]);
    y[i] = v * silu_f(z);
}

// ---------------------------------------------------------------------------
// Host-side launch helpers
// ---------------------------------------------------------------------------
static void launch_gemm(int epi, hipStream_t s,
                        const float* A, const unsigned short* Bw, float* C,
                        const float* bias, int M, int N, int K,
                        int lda, int ldb, int ldc,
                        long long sAb, long long sAh, long long sBb, long long sBh,
                        long long sCb, long long sCh, int batch, int batchH,
                        float alpha) {
    dim3 grid(N / 128, M / 128, batch), block(256);
    switch (epi) {
        case 0: k_gemm<0><<<grid, block, 0, s>>>(A, Bw, C, bias, M, N, K, lda, ldb, ldc,
                    sAb, sAh, sBb, sBh, sCb, sCh, batchH, alpha); break;
        case 1: k_gemm<1><<<grid, block, 0, s>>>(A, Bw, C, bias, M, N, K, lda, ldb, ldc,
                    sAb, sAh, sBb, sBh, sCb, sCh, batchH, alpha); break;
        case 2: k_gemm<2><<<grid, block, 0, s>>>(A, Bw, C, bias, M, N, K, lda, ldb, ldc,
                    sAb, sAh, sBb, sBh, sCb, sCh, batchH, alpha); break;
        default: k_gemm<3><<<grid, block, 0, s>>>(A, Bw, C, bias, M, N, K, lda, ldb, ldc,
                    sAb, sAh, sBb, sBh, sCb, sCh, batchH, alpha); break;
    }
}

static void launch_cvt(hipStream_t s, const float* src, unsigned short* dst, long long n) {
    long long n4 = n / 4;
    int blocks = (int)((n4 + 255) / 256);
    k_cvt_bf16<<<blocks, 256, 0, s>>>(src, dst, n4);
}

// ---------------------------------------------------------------------------
// Orchestration
// ---------------------------------------------------------------------------
extern "C" void kernel_launch(void* const* d_in, const int* in_sizes, int n_in,
                              void* d_out, int out_size, void* d_ws, size_t ws_size,
                              hipStream_t stream) {
    if (n_in < 17) return;
    const int*   idx     = (const int*)d_in[0];
    const float* embed_w = (const float*)d_in[1];
    const float* wq      = (const float*)d_in[2];
    const float* wk      = (const float*)d_in[3];
    const float* wv      = (const float*)d_in[4];
    const float* wo      = (const float*)d_in[5];
    const float* fc1_1   = (const float*)d_in[6];
    const float* fc2_1   = (const float*)d_in[7];
    const float* fc1_2   = (const float*)d_in[8];
    const float* fc2_2   = (const float*)d_in[9];
    const float* in_proj = (const float*)d_in[10];
    const float* out_proj= (const float*)d_in[11];
    const float* conv_w  = (const float*)d_in[12];
    const float* conv_b  = (const float*)d_in[13];
    const float* dt_w    = (const float*)d_in[14];
    const float* dt_b    = (const float*)d_in[15];
    const float* D_param = (const float*)d_in[16];
    float* logits = (float*)d_out;

    // ---- workspace carve-out (256B aligned bump allocator) ----
    char* ws = (char*)d_ws;
    size_t off = 0;
    auto alloc = [&](size_t bytes) -> char* {
        char* p = ws + off;
        off += (bytes + 255) & ~(size_t)255;
        return p;
    };

    // bf16 weights (persist across the launch)
    unsigned short* embed_bf = (unsigned short*)alloc((size_t)VOCAB_ * D_ * 2);
    unsigned short* wq_bf    = (unsigned short*)alloc((size_t)D_ * D_ * 2);
    unsigned short* wk_bf    = (unsigned short*)alloc((size_t)D_ * D_ * 2);
    unsigned short* wv_bf    = (unsigned short*)alloc((size_t)D_ * D_ * 2);
    unsigned short* wo_bf    = (unsigned short*)alloc((size_t)D_ * D_ * 2);
    unsigned short* fc11_bf  = (unsigned short*)alloc((size_t)DFF_ * D_ * 2);
    unsigned short* fc21_bf  = (unsigned short*)alloc((size_t)D_ * DFF_ * 2);
    unsigned short* fc12_bf  = (unsigned short*)alloc((size_t)DFF_ * D_ * 2);
    unsigned short* fc22_bf  = (unsigned short*)alloc((size_t)D_ * DFF_ * 2);
    unsigned short* inp_bf   = (unsigned short*)alloc((size_t)2 * DIN_ * D_ * 2);
    unsigned short* dtw_bf   = (unsigned short*)alloc((size_t)DIN_ * DIN_ * 2);
    unsigned short* outp_bf  = (unsigned short*)alloc((size_t)D_ * DIN_ * 2);

    // fp32 activations
    float* x  = (float*)alloc((size_t)BT_ * D_ * 4);
    float* q  = (float*)alloc((size_t)BT_ * D_ * 4);
    float* k  = (float*)alloc((size_t)BT_ * D_ * 4);
    float* v  = (float*)alloc((size_t)BT_ * D_ * 4);
    float* o  = (float*)alloc((size_t)BT_ * D_ * 4);
    unsigned short* k_bf = (unsigned short*)alloc((size_t)BT_ * D_ * 2);
    unsigned short* vt_bf = (unsigned short*)alloc((size_t)BT_ * D_ * 2);
    // 256MB region reused across phases:
    //   attn: scores[B*H,T,T] | mlp: h[BT,DFF] | mamba: xz[BT,2*DIN]+xconv+dt
    char* Rbase = alloc((size_t)B_ * H_ * T_ * T_ * 4);
    float* scores = (float*)Rbase;
    float* hbuf   = (float*)Rbase;
    float* xz     = (float*)Rbase;
    float* xconv  = (float*)(Rbase + (size_t)BT_ * 2 * DIN_ * 4);
    float* dtbuf  = (float*)(Rbase + (size_t)BT_ * 2 * DIN_ * 4 + (size_t)BT_ * DIN_ * 4);
    float* ybuf   = (float*)alloc((size_t)BT_ * DIN_ * 4);
    (void)ws_size; (void)in_sizes; (void)out_size;

    // ---- 0) weight down-conversion (fp32 -> bf16) ----
    launch_cvt(stream, embed_w, embed_bf, (long long)VOCAB_ * D_);
    launch_cvt(stream, wq, wq_bf, (long long)D_ * D_);
    launch_cvt(stream, wk, wk_bf, (long long)D_ * D_);
    launch_cvt(stream, wv, wv_bf, (long long)D_ * D_);
    launch_cvt(stream, wo, wo_bf, (long long)D_ * D_);
    launch_cvt(stream, fc1_1, fc11_bf, (long long)DFF_ * D_);
    launch_cvt(stream, fc2_1, fc21_bf, (long long)D_ * DFF_);
    launch_cvt(stream, fc1_2, fc12_bf, (long long)DFF_ * D_);
    launch_cvt(stream, fc2_2, fc22_bf, (long long)D_ * DFF_);
    launch_cvt(stream, in_proj, inp_bf, (long long)2 * DIN_ * D_);
    launch_cvt(stream, dt_w, dtw_bf, (long long)DIN_ * DIN_);
    launch_cvt(stream, out_proj, outp_bf, (long long)D_ * DIN_);

    // ---- 1) embedding gather ----
    k_gather<<<BT_, 256, 0, stream>>>(idx, embed_w, x);

    // ---- 2) attention ----
    launch_gemm(0, stream, x, wq_bf, q, nullptr, BT_, D_, D_, D_, D_, D_,
                0, 0, 0, 0, 0, 0, 1, 1, 1.0f);
    launch_gemm(0, stream, x, wk_bf, k, nullptr, BT_, D_, D_, D_, D_, D_,
                0, 0, 0, 0, 0, 0, 1, 1, 1.0f);
    launch_gemm(0, stream, x, wv_bf, v, nullptr, BT_, D_, D_, D_, D_, D_,
                0, 0, 0, 0, 0, 0, 1, 1, 1.0f);
    launch_cvt(stream, k, k_bf, (long long)BT_ * D_);
    {
        long long n = (long long)B_ * H_ * HD_ * T_;
        k_transpose_v<<<(int)((n + 255) / 256), 256, 0, stream>>>(v, vt_bf);
    }
    // scores[bh,q,t'] = alpha * Q . K^T   (batched over b,h)
    launch_gemm(0, stream, q, k_bf, scores, nullptr, T_, T_, HD_, D_, D_, T_,
                (long long)T_ * D_, HD_, (long long)T_ * D_, HD_,
                (long long)H_ * T_ * T_, (long long)T_ * T_,
                B_ * H_, H_, 0.088388347648318447f /* 1/sqrt(128) */);
    {
        dim3 g(T_, B_ * H_);
        k_softmax<<<g, 256, 0, stream>>>(scores);
    }
    // o[b,t,h,hd] = S . V    (batched over b,h; B = V^T bf16)
    launch_gemm(0, stream, scores, vt_bf, o, nullptr, T_, HD_, T_, T_, T_, D_,
                (long long)H_ * T_ * T_, (long long)T_ * T_,
                (long long)H_ * HD_ * T_, (long long)HD_ * T_,
                (long long)T_ * D_, HD_,
                B_ * H_, H_, 1.0f);
    // x += o @ wo^T
    launch_gemm(1, stream, o, wo_bf, x, nullptr, BT_, D_, D_, D_, D_, D_,
                0, 0, 0, 0, 0, 0, 1, 1, 1.0f);

    // ---- 3) MLP 1: x += relu(x@fc1^T)^2 @ fc2^T ----
    launch_gemm(2, stream, x, fc11_bf, hbuf, nullptr, BT_, DFF_, D_, D_, D_, DFF_,
                0, 0, 0, 0, 0, 0, 1, 1, 1.0f);
    launch_gemm(1, stream, hbuf, fc21_bf, x, nullptr, BT_, D_, DFF_, DFF_, DFF_, D_,
                0, 0, 0, 0, 0, 0, 1, 1, 1.0f);

    // ---- 4) mamba ----
    launch_gemm(0, stream, x, inp_bf, xz, nullptr, BT_, 2 * DIN_, D_, D_, D_, 2 * DIN_,
                0, 0, 0, 0, 0, 0, 1, 1, 1.0f);
    {
        long long n = (long long)BT_ * DIN_;
        int blocks = (int)((n + 255) / 256);
        k_conv_silu<<<blocks, 256, 0, stream>>>(xz, conv_w, conv_b, xconv);
        launch_gemm(3, stream, xconv, dtw_bf, dtbuf, dt_b, BT_, DIN_, DIN_,
                    DIN_, DIN_, DIN_, 0, 0, 0, 0, 0, 0, 1, 1, 1.0f);
        k_mamba_mix<<<blocks, 256, 0, stream>>>(xconv, dtbuf, xz, D_param, ybuf);
    }
    launch_gemm(1, stream, ybuf, outp_bf, x, nullptr, BT_, D_, DIN_, DIN_, DIN_, D_,
                0, 0, 0, 0, 0, 0, 1, 1, 1.0f);

    // ---- 5) MLP 2 ----
    launch_gemm(2, stream, x, fc12_bf, hbuf, nullptr, BT_, DFF_, D_, D_, D_, DFF_,
                0, 0, 0, 0, 0, 0, 1, 1, 1.0f);
    launch_gemm(1, stream, hbuf, fc22_bf, x, nullptr, BT_, D_, DFF_, DFF_, DFF_, D_,
                0, 0, 0, 0, 0, 0, 1, 1, 1.0f);

    // ---- 6) tied lm_head: logits = x @ embed^T ----
    launch_gemm(0, stream, x, embed_bf, logits, nullptr, BT_, VOCAB_, D_,
                D_, D_, VOCAB_, 0, 0, 0, 0, 0, 0, 1, 1, 1.0f);
}
